// FusionPhysMamba_V02_52733608461137
// MI455X (gfx1250) — compile-verified, matrix operand order checked by reading, not run on hardware
//
#include <hip/hip_runtime.h>
#include <math.h>

typedef __attribute__((ext_vector_type(16))) _Float16 v16h;
typedef __attribute__((ext_vector_type(8)))  _Float16 v8h;
typedef __attribute__((ext_vector_type(8)))  float    v8f;

#define DIMC    128
#define D_INNER 128
#define D_STATE 8
#define DT_RANK 8
#define D_CONV  4
#define BATCH   4
#define NTOK    16384               // L*H*W = 256*8*8
#define MTOT    (BATCH * NTOK)      // 65536 tokens

// ---------------------------------------------------------------------------
// f32 -> f16 weight conversion WITH swizzle into WMMA B-fragment order.
// Fragment layout (halves): dst[((k0Idx*128 + col)*2 + sel)*16 + j]
//   j<8  -> W[k0Idx*32 + sel*8 + j     ][col]
//   j>=8 -> W[k0Idx*32 + sel*8 + j + 8 ][col]
// so each consuming lane's 16 B halves are one contiguous 32-byte LDS chunk.
// ---------------------------------------------------------------------------
__global__ void cvt_swizzle_w_kernel(const float* __restrict__ src,
                                     _Float16* __restrict__ dst, int nHalves) {
    int i = blockIdx.x * blockDim.x + threadIdx.x;
    if (i >= nHalves) return;
    int mat   = i >> 14;            // which 128x128 matrix
    int r     = i & 16383;
    int j     = r & 15;             // half index within fragment chunk
    int chunk = r >> 4;             // (k0Idx*128 + col)*2 + sel
    int sel   = chunk & 1;
    int colk  = chunk >> 1;
    int col   = colk & 127;
    int k0Idx = colk >> 7;          // 0..3
    int koff  = j + ((j >= 8) ? 8 : 0);
    int k     = k0Idx * 32 + sel * 8 + koff;
    dst[i] = (_Float16)src[(size_t)mat * 16384 + k * 128 + col];
}

// ---------------------------------------------------------------------------
// Fused transpose (B,DIM,N)->(B,N,DIM) + LayerNorm + cast to f16.
// One wave (32 lanes) per token; each lane owns 4 channels.
// ---------------------------------------------------------------------------
__global__ __launch_bounds__(256)
void ln_transpose_f16_kernel(const float* __restrict__ x,     // (B, DIM, N)
                             const float* __restrict__ w,     // (DIM)
                             const float* __restrict__ bias,  // (DIM)
                             _Float16* __restrict__ out) {    // (B, N, DIM)
    int wave = (blockIdx.x * blockDim.x + threadIdx.x) >> 5;
    int lane = threadIdx.x & 31;
    int b = wave >> 14;            // / NTOK
    int n = wave & (NTOK - 1);
    const float* xb = x + (size_t)b * DIMC * NTOK + n;
    float v0 = xb[(size_t)(lane      ) * NTOK];
    float v1 = xb[(size_t)(lane + 32 ) * NTOK];
    float v2 = xb[(size_t)(lane + 64 ) * NTOK];
    float v3 = xb[(size_t)(lane + 96 ) * NTOK];
    float s = v0 + v1 + v2 + v3;
    float q = v0 * v0 + v1 * v1 + v2 * v2 + v3 * v3;
    #pragma unroll
    for (int off = 16; off > 0; off >>= 1) {
        s += __shfl_xor(s, off, 32);
        q += __shfl_xor(q, off, 32);
    }
    float mean = s * (1.0f / DIMC);
    float var  = q * (1.0f / DIMC) - mean * mean;
    float rstd = rsqrtf(var + 1e-5f);
    _Float16* ob = out + (size_t)(b * NTOK + n) * DIMC;
    ob[lane     ] = (_Float16)((v0 - mean) * rstd * w[lane     ] + bias[lane     ]);
    ob[lane + 32] = (_Float16)((v1 - mean) * rstd * w[lane + 32] + bias[lane + 32]);
    ob[lane + 64] = (_Float16)((v2 - mean) * rstd * w[lane + 64] + bias[lane + 64]);
    ob[lane + 96] = (_Float16)((v3 - mean) * rstd * w[lane + 96] + bias[lane + 96]);
}

// ---------------------------------------------------------------------------
// A-fragment loader: 16x32 f16 A layout = two contiguous 16B chunks per lane
// (K groups at kSel and kSel+16).
// ---------------------------------------------------------------------------
__device__ __forceinline__ v16h load_a_frag(const _Float16* Ar) {
    v8h lo = *(const v8h*)(Ar);        // K = kSel .. kSel+7
    v8h hi = *(const v8h*)(Ar + 16);   // K = kSel+16 .. kSel+23
    v16h af;
    #pragma unroll
    for (int j = 0; j < 8; ++j) { af[j] = lo[j]; af[j + 8] = hi[j]; }
    return af;
}

// ---------------------------------------------------------------------------
// WMMA GEMM: C(M x 128) = A(M x 128, f16) * W(128 x 128, f16) [+ C] [+ bias]
// W pre-swizzled in B-fragment order, staged in LDS. Each wave owns a 64x16
// output tile (4 accumulators); one B fragment reused across 4 WMMAs per K
// step -> 16 v_wmma per wave. K loop software-pipelined: fragments for step
// k+1 are issued before the WMMAs of step k so loads overlap the matrix pipe.
// flags bit0 = accumulate into C, bit1 = store transposed to (B, DIM, NTOK).
// ---------------------------------------------------------------------------
__global__ __launch_bounds__(256)
void gemm128_wmma_kernel(const _Float16* __restrict__ A,
                         const _Float16* __restrict__ Wf,   // swizzled
                         const float* __restrict__ bias,
                         float* __restrict__ C,
                         int flags) {
    __shared__ __align__(64) _Float16 sW[128 * 128];   // 32 KB, fragment order
    {
        const uint4* src = (const uint4*)Wf;
        uint4*       dst = (uint4*)sW;
        for (int i = threadIdx.x; i < (128 * 128) / 8; i += 256) dst[i] = src[i];
    }
    __syncthreads();

    int waveId  = threadIdx.x >> 5;
    int lane    = threadIdx.x & 31;
    int rowBase = blockIdx.x * 64;       // 64 token rows per block (4 tiles/wave)
    int colBase = waveId * 16;           // 8 waves cover 128 output cols
    int col     = colBase + (lane & 15);
    int sel     = lane >> 4;             // 0 / 1
    int kSel    = sel << 3;              // 0 / 8
    int rOff    = sel << 3;              // C fragment: M = r + 8*(lane/16)

    v8f acc0 = {}, acc1 = {}, acc2 = {}, acc3 = {};
    if (flags & 1) {
        #pragma unroll
        for (int r = 0; r < 8; ++r) {
            acc0[r] = C[(size_t)(rowBase +  0 + r + rOff) * 128 + col];
            acc1[r] = C[(size_t)(rowBase + 16 + r + rOff) * 128 + col];
            acc2[r] = C[(size_t)(rowBase + 32 + r + rOff) * 128 + col];
            acc3[r] = C[(size_t)(rowBase + 48 + r + rOff) * 128 + col];
        }
    }

    const _Float16* A0 = A + (size_t)(rowBase + (lane & 15)) * 128 + kSel;
    // prologue: fragments for k0Idx = 0 (tile stride = 16 rows * 128 = 2048)
    v16h bf = *(const v16h*)&sW[((0 * 128 + col) * 2 + sel) * 16];
    v16h a0 = load_a_frag(A0 + 0 * 2048);
    v16h a1 = load_a_frag(A0 + 1 * 2048);
    v16h a2 = load_a_frag(A0 + 2 * 2048);
    v16h a3 = load_a_frag(A0 + 3 * 2048);

    #pragma unroll
    for (int k0Idx = 0; k0Idx < 4; ++k0Idx) {
        v16h bfn = {}, a0n = {}, a1n = {}, a2n = {}, a3n = {};
        if (k0Idx < 3) {
            const _Float16* An = A0 + (k0Idx + 1) * 32;
            bfn = *(const v16h*)&sW[(((k0Idx + 1) * 128 + col) * 2 + sel) * 16];
            a0n = load_a_frag(An + 0 * 2048);
            a1n = load_a_frag(An + 1 * 2048);
            a2n = load_a_frag(An + 2 * 2048);
            a3n = load_a_frag(An + 3 * 2048);
        }
        acc0 = __builtin_amdgcn_wmma_f32_16x16x32_f16(false, a0, false, bf, (short)0, acc0, false, false);
        acc1 = __builtin_amdgcn_wmma_f32_16x16x32_f16(false, a1, false, bf, (short)0, acc1, false, false);
        acc2 = __builtin_amdgcn_wmma_f32_16x16x32_f16(false, a2, false, bf, (short)0, acc2, false, false);
        acc3 = __builtin_amdgcn_wmma_f32_16x16x32_f16(false, a3, false, bf, (short)0, acc3, false, false);
        if (k0Idx < 3) { bf = bfn; a0 = a0n; a1 = a1n; a2 = a2n; a3 = a3n; }
    }

    float bv = bias ? bias[col] : 0.0f;
    #pragma unroll
    for (int t = 0; t < 4; ++t) {
        v8f acc = (t == 0) ? acc0 : (t == 1) ? acc1 : (t == 2) ? acc2 : acc3;
        if (flags & 2) {
            // transposed store into (B, DIM, NTOK)
            #pragma unroll
            for (int r = 0; r < 8; ++r) {
                int row = rowBase + t * 16 + r + rOff;   // global token index
                int b = row >> 14, n = row & (NTOK - 1);
                C[((size_t)b * DIMC + col) * NTOK + n] = acc[r] + bv;
            }
        } else {
            #pragma unroll
            for (int r = 0; r < 8; ++r)
                C[(size_t)(rowBase + t * 16 + r + rOff) * 128 + col] = acc[r] + bv;
        }
    }
}

// ---------------------------------------------------------------------------
// Causal depthwise conv1d (kernel 4) along tokens + bias + SiLU
// ---------------------------------------------------------------------------
__global__ void conv_silu_kernel(const float* __restrict__ u,   // (B,N,128)
                                 const float* __restrict__ cw,  // (128,4)
                                 const float* __restrict__ cb,  // (128)
                                 float* __restrict__ uc) {
    int idx = blockIdx.x * blockDim.x + threadIdx.x;   // over MTOT*128
    int d = idx & 127;
    int t = idx >> 7;              // b*N + n
    int n = t & (NTOK - 1);
    float acc = cb[d];
    #pragma unroll
    for (int k = 0; k < D_CONV; ++k) {
        int nn = n + k - (D_CONV - 1);
        if (nn >= 0) acc += u[(size_t)(t + k - (D_CONV - 1)) * 128 + d] * cw[d * D_CONV + k];
    }
    uc[idx] = acc / (1.0f + __expf(-acc));             // SiLU
}

// ---------------------------------------------------------------------------
// x_dbl = uc @ Wxproj -> (dt_r | B | C); dt = softplus(dt_r @ Wdt + bdt)
// One 128-thread block per token (tiny GEMMs, VALU is fine).
// ---------------------------------------------------------------------------
__global__ __launch_bounds__(128)
void xproj_dt_kernel(const float* __restrict__ uc,      // (M,128)
                     const float* __restrict__ Wxproj,  // (128,24)
                     const float* __restrict__ Wdt,     // (8,128)
                     const float* __restrict__ bdt,     // (128)
                     float* __restrict__ dt,            // (M,128)
                     float* __restrict__ Bm,            // (M,8)
                     float* __restrict__ Cm) {          // (M,8)
    __shared__ float s_dtr[DT_RANK];
    int t = blockIdx.x;
    int j = threadIdx.x;
    const float* ur = uc + (size_t)t * 128;
    if (j < DT_RANK + 2 * D_STATE) {
        float a = 0.0f;
        for (int c = 0; c < 128; ++c) a += ur[c] * Wxproj[c * 24 + j];
        if (j < DT_RANK)                s_dtr[j] = a;
        else if (j < DT_RANK + D_STATE) Bm[(size_t)t * 8 + (j - DT_RANK)] = a;
        else                            Cm[(size_t)t * 8 + (j - DT_RANK - D_STATE)] = a;
    }
    __syncthreads();
    float a = bdt[j];
    #pragma unroll
    for (int r = 0; r < DT_RANK; ++r) a += s_dtr[r] * Wdt[r * 128 + j];
    dt[(size_t)t * 128 + j] = (a > 20.0f) ? a : log1pf(__expf(a));   // softplus
}

// ---------------------------------------------------------------------------
// Serial SSM scan. One workgroup per batch; thread = (d, s); shfl_xor reduce
// over the 8 D_STATE lanes; fuses Dp skip and SiLU(z) gate; emits f16.
// ---------------------------------------------------------------------------
__global__ __launch_bounds__(1024)
void scan_kernel(const float* __restrict__ dt,    // (B,N,128)
                 const float* __restrict__ uc,    // (B,N,128)
                 const float* __restrict__ zg,    // (B,N,128) pre-activation z
                 const float* __restrict__ Bm,    // (B,N,8)
                 const float* __restrict__ Cm,    // (B,N,8)
                 const float* __restrict__ A_log, // (128,8)
                 const float* __restrict__ Dp,    // (128)
                 _Float16* __restrict__ yg) {     // (B,N,128)
    int b   = blockIdx.x;
    int tid = threadIdx.x;
    int d = tid >> 3;
    int s = tid & 7;
    float Ads = -__expf(A_log[d * 8 + s]);
    float Dd  = Dp[d];
    float h   = 0.0f;
    size_t base = (size_t)b * NTOK;
    for (int n = 0; n < NTOK; ++n) {
        size_t t = base + n;
        float dtv = dt[t * 128 + d];
        float ucv = uc[t * 128 + d];
        float Bv  = Bm[t * 8 + s];
        float Cv  = Cm[t * 8 + s];
        h = __expf(dtv * Ads) * h + dtv * Bv * ucv;
        float y = h * Cv;
        y += __shfl_xor(y, 1, 32);
        y += __shfl_xor(y, 2, 32);
        y += __shfl_xor(y, 4, 32);
        if (s == 0) {
            float zv = zg[t * 128 + d];
            float g  = zv / (1.0f + __expf(-zv));
            yg[t * 128 + d] = (_Float16)((y + Dd * ucv) * g);
        }
    }
}

// ---------------------------------------------------------------------------
// s = 0.5*(m0 + m1 + ff + bf)  (transposing the raw inputs), cast f16
// ---------------------------------------------------------------------------
__global__ void final_prep_kernel(const float* __restrict__ msum,  // (B,N,128)
                                  const float* __restrict__ front, // (B,128,N)
                                  const float* __restrict__ back,  // (B,128,N)
                                  _Float16* __restrict__ sh) {     // (B,N,128)
    int idx = blockIdx.x * blockDim.x + threadIdx.x;
    int c = idx & 127;
    int t = idx >> 7;
    int b = t >> 14;
    int n = t & (NTOK - 1);
    size_t src = ((size_t)b * DIMC + c) * NTOK + n;
    sh[idx] = (_Float16)(0.5f * (msum[idx] + front[src] + back[src]));
}

// ---------------------------------------------------------------------------
// Host-side pipeline
// ---------------------------------------------------------------------------
extern "C" void kernel_launch(void* const* d_in, const int* in_sizes, int n_in,
                              void* d_out, int out_size, void* d_ws, size_t ws_size,
                              hipStream_t stream) {
    const float* front   = (const float*)d_in[0];
    const float* back    = (const float*)d_in[1];
    const float* ln0_w   = (const float*)d_in[2];
    const float* ln0_b   = (const float*)d_in[3];
    const float* ln1_w   = (const float*)d_in[4];
    const float* ln1_b   = (const float*)d_in[5];
    const float* Wx      = (const float*)d_in[6];
    const float* Wz      = (const float*)d_in[7];
    const float* conv_w  = (const float*)d_in[8];
    const float* conv_b  = (const float*)d_in[9];
    const float* Wxproj  = (const float*)d_in[10];
    const float* Wdt     = (const float*)d_in[11];
    const float* bdt     = (const float*)d_in[12];
    const float* A_log   = (const float*)d_in[13];
    const float* Dp      = (const float*)d_in[14];
    const float* Wout_m  = (const float*)d_in[15];
    const float* bout_m  = (const float*)d_in[16];
    const float* Wout    = (const float*)d_in[17];
    const float* bout    = (const float*)d_in[18];

    char* ws = (char*)d_ws;
    // ---- static workspace layout (bytes) ----
    _Float16* WXh  = (_Float16*)(ws);                       // 2*16384 halves
    _Float16* WZh  = WXh  + 2 * 16384;
    _Float16* WOMh = WZh  + 2 * 16384;
    _Float16* WOh  = WOMh + 2 * 16384;                      // weights < 256 KB
    size_t off = 0x40000;
    _Float16* AXh  = (_Float16*)(ws + off); off += (size_t)MTOT * 128 * 2;  // 16 MB
    _Float16* AZh  = (_Float16*)(ws + off); off += (size_t)MTOT * 128 * 2;  // 16 MB
    float*    U    = (float*)(ws + off);    off += (size_t)MTOT * 128 * 4;  // 32 MB (also dt)
    float*    Z    = (float*)(ws + off);    off += (size_t)MTOT * 128 * 4;  // 32 MB
    float*    UC   = (float*)(ws + off);    off += (size_t)MTOT * 128 * 4;  // 32 MB
    _Float16* YGh  = (_Float16*)(ws + off); off += (size_t)MTOT * 128 * 2;  // 16 MB (also s_h)
    float*    BM   = (float*)(ws + off);    off += (size_t)MTOT * 8 * 4;    // 2 MB
    float*    CMb  = (float*)(ws + off);    off += (size_t)MTOT * 8 * 4;    // 2 MB
    float*    MS   = (float*)(ws + off);    off += (size_t)MTOT * 128 * 4;  // 32 MB
    float*    DT   = U;       // alias: u consumed by conv before dt written
    _Float16* SH   = YGh;     // alias: yg consumed by GEMM3 before s written

    // 1) weight conversions to f16 (swizzled into WMMA B-fragment order)
    cvt_swizzle_w_kernel<<<(2 * 16384 + 255) / 256, 256, 0, stream>>>(Wx, WXh, 2 * 16384);
    cvt_swizzle_w_kernel<<<(2 * 16384 + 255) / 256, 256, 0, stream>>>(Wz, WZh, 2 * 16384);
    cvt_swizzle_w_kernel<<<(2 * 16384 + 255) / 256, 256, 0, stream>>>(Wout_m, WOMh, 2 * 16384);
    cvt_swizzle_w_kernel<<<(16384 + 255) / 256, 256, 0, stream>>>(Wout, WOh, 16384);

    // 2) two cross-mamba blocks (sequential; buffers reused)
    for (int i = 0; i < 2; ++i) {
        const float* xsrc = (i == 0) ? front : back;
        const float* zsrc = (i == 0) ? back  : front;

        ln_transpose_f16_kernel<<<MTOT / 8, 256, 0, stream>>>(
            xsrc, ln0_w + i * DIMC, ln0_b + i * DIMC, AXh);
        ln_transpose_f16_kernel<<<MTOT / 8, 256, 0, stream>>>(
            zsrc, ln1_w + i * DIMC, ln1_b + i * DIMC, AZh);

        gemm128_wmma_kernel<<<MTOT / 64, 256, 0, stream>>>(AXh, WXh + i * 16384, nullptr, U, 0);
        gemm128_wmma_kernel<<<MTOT / 64, 256, 0, stream>>>(AZh, WZh + i * 16384, nullptr, Z, 0);

        conv_silu_kernel<<<(MTOT * 128) / 256, 256, 0, stream>>>(
            U, conv_w + i * DIMC * D_CONV, conv_b + i * DIMC, UC);

        xproj_dt_kernel<<<MTOT, 128, 0, stream>>>(
            UC, Wxproj + i * 128 * 24, Wdt + i * 8 * 128, bdt + i * 128, DT, BM, CMb);

        scan_kernel<<<BATCH, 1024, 0, stream>>>(
            DT, UC, Z, BM, CMb, A_log + i * 128 * 8, Dp + i * 128, YGh);

        gemm128_wmma_kernel<<<MTOT / 64, 256, 0, stream>>>(
            YGh, WOMh + i * 16384, bout_m + i * DIMC, MS, (i == 0) ? 0 : 1);
    }

    // 3) residual add + 0.5 scale + final GEMM, stored transposed to (B,DIM,N)
    final_prep_kernel<<<(MTOT * 128) / 256, 256, 0, stream>>>(MS, front, back, SH);
    gemm128_wmma_kernel<<<MTOT / 64, 256, 0, stream>>>(SH, WOh, bout, (float*)d_out, 2);
}
